// CoverTreeLoss_5823975653575
// MI455X (gfx1250) — matrix operand order repaired and checked by reading, not compiled, not run on hardware
//
#include <hip/hip_runtime.h>
#include <hip/hip_bf16.h>

// ---------------------------------------------------------------------------
// CoverTreeLoss on gfx1250:
//   1) split x into bf16 hi/lo                       (elementwise)
//   2) added_weights = weights[k] + path gather-sum  (1 wave / class)
//      + emit fp32 to d_out and bf16 hi/lo to ws
//   3) logits = x @ added_weights^T via v_wmma_f32_16x16x32_bf16,
//      3-term bf16 split (hi*hi + hi*lo + lo*hi), fp32 accumulate.
//      WMMAs issued term-major across 4 independent accumulators so
//      dependent WMMAs are >=4 instructions apart (no v_nop bubbles).
//   4) per-row online logsumexp + CE term            (1 block / row)
//   5) mean over rows -> loss                        (1 block)
// ---------------------------------------------------------------------------

#define LENGTH   65536
#define DDIM     128
#define BATCH    1024
#define KCLS     50000
#define MAXDEPTH 16

typedef __attribute__((ext_vector_type(16))) __bf16 v16bf;
typedef __attribute__((ext_vector_type(8)))  float  v8f;

union FragU { v16bf v; uint4 u[2]; };

// ---- A fragment: 16x32 bf16 tile, per-lane chunks at +0 and +16 elements --
// (lanes 0-15: row m0+lane, K {0..7,16..23}; lanes 16-31: K {8..15,24..31})
__device__ __forceinline__ v16bf load_a_frag(const __bf16* p) {
    FragU f;
    f.u[0] = *(const uint4*)(p);
    f.u[1] = *(const uint4*)(p + 16);
    return f.v;
}

// ---- B fragment: 32x16 bf16 tile, per-lane chunks at +0 and +16 elements --
// (lanes 0-15: col n0+lane, K {0..7,16..23}; lanes 16-31: K {8..15,24..31})
__device__ __forceinline__ v16bf load_b_frag(const __bf16* p) {
    FragU f;
    f.u[0] = *(const uint4*)(p);
    f.u[1] = *(const uint4*)(p + 16);
    return f.v;
}

// ---------------------------------------------------------------------------
// Kernel 0: split x (fp32) into bf16 hi + bf16 residual lo
__global__ __launch_bounds__(256)
void split_x_kernel(const float* __restrict__ x, __bf16* __restrict__ xhi,
                    __bf16* __restrict__ xlo) {
    int i = blockIdx.x * 256 + threadIdx.x;
    if (i >= BATCH * DDIM) return;
    float v = x[i];
    __bf16 h = (__bf16)v;
    xhi[i] = h;
    xlo[i] = (__bf16)(v - (float)h);
}

// ---------------------------------------------------------------------------
// Kernel 1: added_weights[k] = weights[k] + sum of ancestor rows.
// One wave per class; lane handles 4 consecutive floats (float4 = full row / wave).
__global__ __launch_bounds__(256)
void added_weights_kernel(const float* __restrict__ weights,
                          const long long* __restrict__ path_idx,
                          const int* __restrict__ path_len,
                          float* __restrict__ aw_out,      // d_out tail (4B aligned only)
                          __bf16* __restrict__ aw_hi,
                          __bf16* __restrict__ aw_lo) {
    int wave = threadIdx.x >> 5;
    int lane = threadIdx.x & 31;
    int k = blockIdx.x * 8 + wave;
    if (k >= KCLS) return;                       // wave-uniform
    int d = lane * 4;
    float4 acc = *(const float4*)(weights + (size_t)k * DDIM + d);
    int len = path_len[k];
    const long long* pix = path_idx + (size_t)k * MAXDEPTH;
    for (int p = 0; p < len; ++p) {
        long long idx = pix[p];
        float4 wv = *(const float4*)(weights + (size_t)idx * DDIM + d);
        acc.x += wv.x; acc.y += wv.y; acc.z += wv.z; acc.w += wv.w;
    }
    float vals[4] = {acc.x, acc.y, acc.z, acc.w};
    size_t base = (size_t)k * DDIM + d;
#pragma unroll
    for (int i = 0; i < 4; ++i) {
        float v = vals[i];
        aw_out[base + i] = v;                    // fp32 output (scalar: 4B aligned)
        __bf16 h = (__bf16)v;
        aw_hi[base + i] = h;
        aw_lo[base + i] = (__bf16)(v - (float)h);
    }
}

// ---------------------------------------------------------------------------
// Kernel 2: logits[BATCH, KCLS] = X @ AW^T with split-bf16 WMMA.
// Wave computes a 16(M) x 64(N) strip: 4 accumulators, A regs reused 4x.
// wave id -> (m_tile = wid & 63, n_group = wid >> 6) so the 64 M-tiles of one
// N-group run on adjacent waves (8 consecutive blocks) => B tiles hit in L2.
#define NGROUPS 782   // ceil(3125 N-subtiles / 4)
__global__ __launch_bounds__(256)
void gemm_kernel(const __bf16* __restrict__ xhi, const __bf16* __restrict__ xlo,
                 const __bf16* __restrict__ ahi, const __bf16* __restrict__ alo,
                 float* __restrict__ logits) {
    const int lane = threadIdx.x & 31;
    const long wid = (long)blockIdx.x * 8 + (threadIdx.x >> 5);
    const int m0     = (int)(wid & 63) * 16;
    const int ngrp   = (int)(wid >> 6);
    const int n_base = ngrp * 64;

    // K-invariant per-lane offsets (clamped rows keep EXEC all-ones; stores guarded)
    const size_t a_off = (size_t)(m0 + (lane & 15)) * DDIM + (lane >> 4) * 8;
    const __bf16* pAh = xhi + a_off;
    const __bf16* pAl = xlo + a_off;

    size_t b_off[4];
#pragma unroll
    for (int t = 0; t < 4; ++t) {
        int n = n_base + t * 16 + (lane & 15);
        if (n >= KCLS) n = KCLS - 1;
        b_off[t] = (size_t)n * DDIM + (lane >> 4) * 8;
    }

    v8f c[4];
#pragma unroll
    for (int t = 0; t < 4; ++t) c[t] = (v8f){0.f,0.f,0.f,0.f,0.f,0.f,0.f,0.f};

#pragma unroll
    for (int k0 = 0; k0 < DDIM; k0 += 32) {
        v16bf a_hi = load_a_frag(pAh + k0);
        v16bf a_lo = load_a_frag(pAl + k0);
        v16bf bh[4], bl[4];
#pragma unroll
        for (int t = 0; t < 4; ++t) {
            bh[t] = load_b_frag(ahi + b_off[t] + k0);
            bl[t] = load_b_frag(alo + b_off[t] + k0);
        }
        // term-major issue: dependent WMMAs (same c[t]) are 4 apart
#pragma unroll
        for (int t = 0; t < 4; ++t)
            c[t] = __builtin_amdgcn_wmma_f32_16x16x32_bf16(
                       false, a_hi, false, bh[t], (short)0, c[t], false, false);
#pragma unroll
        for (int t = 0; t < 4; ++t)
            c[t] = __builtin_amdgcn_wmma_f32_16x16x32_bf16(
                       false, a_hi, false, bl[t], (short)0, c[t], false, false);
#pragma unroll
        for (int t = 0; t < 4; ++t)
            c[t] = __builtin_amdgcn_wmma_f32_16x16x32_bf16(
                       false, a_lo, false, bh[t], (short)0, c[t], false, false);
    }

    // C layout: VGPR r -> row m0 + r + 8*(lane>=16), col n0 + (lane&15)
    const int n_off  = lane & 15;
    const int m_half = (lane >> 4) * 8;
#pragma unroll
    for (int t = 0; t < 4; ++t) {
        int n0 = n_base + t * 16;
        if (n0 >= KCLS) continue;               // subtiles are whole-valid or whole-invalid
#pragma unroll
        for (int r = 0; r < 8; ++r) {
            int m = m0 + m_half + r;
            logits[(size_t)m * KCLS + n0 + n_off] = c[t][r];
        }
    }
}

// ---------------------------------------------------------------------------
// Kernel 3: per-row online logsumexp + CE term: row_out[i] = logZ_i - logits[i, y_i]
__global__ __launch_bounds__(256)
void lse_kernel(const float* __restrict__ logits, const long long* __restrict__ y,
                float* __restrict__ row_out) {
    const int i = blockIdx.x;
    const int t = threadIdx.x;
    const float* row = logits + (size_t)i * KCLS;
    float m = -__builtin_inff();
    float s = 0.f;
    for (int j = t; j < KCLS; j += 256) {
        float v = row[j];
        if (v > m) { s = s * __expf(m - v) + 1.0f; m = v; }
        else       { s += __expf(v - m); }
    }
    __shared__ float sm[256], ss[256];
    sm[t] = m; ss[t] = s;
    __syncthreads();
    for (int off = 128; off > 0; off >>= 1) {
        if (t < off) {
            float m2 = sm[t + off], s2 = ss[t + off];
            float mm = fmaxf(sm[t], m2);
            ss[t] = ss[t] * __expf(sm[t] - mm) + s2 * __expf(m2 - mm);
            sm[t] = mm;
        }
        __syncthreads();
    }
    if (t == 0) {
        float ly = row[(size_t)y[i]];
        row_out[i] = logf(ss[0]) + sm[0] - ly;   // -logp[i, y_i]
    }
}

// Kernel 4: loss = mean(row_out)
__global__ __launch_bounds__(256)
void loss_kernel(const float* __restrict__ row_out, float* __restrict__ loss) {
    __shared__ float sbuf[256];
    float a = 0.f;
    for (int i = threadIdx.x; i < BATCH; i += 256) a += row_out[i];
    sbuf[threadIdx.x] = a;
    __syncthreads();
    for (int off = 128; off > 0; off >>= 1) {
        if (threadIdx.x < off) sbuf[threadIdx.x] += sbuf[threadIdx.x + off];
        __syncthreads();
    }
    if (threadIdx.x == 0) loss[0] = sbuf[0] / (float)BATCH;
}

// ---------------------------------------------------------------------------
extern "C" void kernel_launch(void* const* d_in, const int* in_sizes, int n_in,
                              void* d_out, int out_size, void* d_ws, size_t ws_size,
                              hipStream_t stream) {
    const float*     weights  = (const float*)d_in[0];
    const float*     x        = (const float*)d_in[1];
    const long long* y        = (const long long*)d_in[2];
    const long long* path_idx = (const long long*)d_in[3];
    const int*       path_len = (const int*)d_in[4];

    float* out    = (float*)d_out;
    float* loss   = out;                                   // [1]
    float* logits = out + 1;                               // [BATCH*KCLS]
    float* aw     = out + 1 + (size_t)BATCH * KCLS;        // [KCLS*DDIM]

    char* ws = (char*)d_ws;
    const size_t XB  = (size_t)BATCH * DDIM * 2;           // 256 KB
    const size_t AWB = (size_t)KCLS  * DDIM * 2;           // 12.8 MB
    __bf16* xhi    = (__bf16*)(ws);
    __bf16* xlo    = (__bf16*)(ws + XB);
    __bf16* aw_hi  = (__bf16*)(ws + 2 * XB);
    __bf16* aw_lo  = (__bf16*)(ws + 2 * XB + AWB);
    float*  rowbuf = (float*) (ws + 2 * XB + 2 * AWB);     // [BATCH]

    // 0) split x
    split_x_kernel<<<(BATCH * DDIM + 255) / 256, 256, 0, stream>>>(x, xhi, xlo);

    // 1) added_weights (1 wave / class), 8 waves per block
    added_weights_kernel<<<(KCLS + 7) / 8, 256, 0, stream>>>(
        weights, path_idx, path_len, aw, aw_hi, aw_lo);

    // 2) GEMM: 64 m-tiles * NGROUPS n-groups wave-jobs, 8 waves / block
    const int wave_jobs = 64 * NGROUPS;                    // 50048
    gemm_kernel<<<wave_jobs / 8, 256, 0, stream>>>(xhi, xlo, aw_hi, aw_lo, logits);

    // 3) per-row logsumexp + CE term
    lse_kernel<<<BATCH, 256, 0, stream>>>(logits, y, rowbuf);

    // 4) mean -> loss
    loss_kernel<<<1, 256, 0, stream>>>(rowbuf, loss);
}
